// NeuronEquivDeepSetLayer_translation_11922829214364
// MI455X (gfx1250) — matrix operand order repaired and checked by reading, not compiled, not run on hardware
//
#include <hip/hip_runtime.h>

// ---------------- problem constants ----------------
#define NROWS      1000000
#define NTILES     62500            // NROWS / 16
#define DD         64
#define NLAYERS    16
#define NSEG       2048             // 128 graphs * 16 layers
#define REPL       8                // replicated atomic accumulators

typedef __attribute__((ext_vector_type(16))) _Float16 v16h;
typedef __attribute__((ext_vector_type(8)))  _Float16 v8h;
typedef __attribute__((ext_vector_type(8)))  float    v8f;

// ---------------- small helpers ----------------
__device__ __forceinline__ float4 f4add(float4 a, float4 b) {
  return make_float4(a.x + b.x, a.y + b.y, a.z + b.z, a.w + b.w);
}

// Pack 16 f32 (two K-halves of 8) into a 16-bit A/B fragment register set.
__device__ __forceinline__ v16h pack_frag(float4 f0, float4 f1, float4 f2, float4 f3) {
  v16h a;
  a[0]  = (_Float16)f0.x; a[1]  = (_Float16)f0.y; a[2]  = (_Float16)f0.z; a[3]  = (_Float16)f0.w;
  a[4]  = (_Float16)f1.x; a[5]  = (_Float16)f1.y; a[6]  = (_Float16)f1.z; a[7]  = (_Float16)f1.w;
  a[8]  = (_Float16)f2.x; a[9]  = (_Float16)f2.y; a[10] = (_Float16)f2.z; a[11] = (_Float16)f2.w;
  a[12] = (_Float16)f3.x; a[13] = (_Float16)f3.y; a[14] = (_Float16)f3.z; a[15] = (_Float16)f3.w;
  return a;
}

// Load A-fragment (16x32 f16, ISA 7.12.2 layout) from row-major f32 at p.
// p must already point at x[row][k0 + hi*8].
__device__ __forceinline__ v16h load_a_frag_f32(const float* __restrict__ p) {
  float4 f0 = *(const float4*)(p + 0);    // K +0..3
  float4 f1 = *(const float4*)(p + 4);    // K +4..7
  float4 f2 = *(const float4*)(p + 16);   // K +16..19
  float4 f3 = *(const float4*)(p + 20);   // K +20..23
  return pack_frag(f0, f1, f2, f3);
}

// Stage a 64x64 row-major f32 weight matrix into LDS as TRANSPOSED f16:
// Wt[n*64 + k] = W[k*64 + n].  256 threads cooperate.
__device__ __forceinline__ void stage_wt(const float* __restrict__ W, _Float16* Wt) {
  int t = threadIdx.x;
#pragma unroll
  for (int j = 0; j < 4; ++j) {
    int base = t * 16 + j * 4;           // 0..4092, steps of 4 within a row
    float4 v = *(const float4*)(W + base);
    int k = base >> 6;
    int n = base & 63;
    Wt[(n + 0) * 64 + k] = (_Float16)v.x;
    Wt[(n + 1) * 64 + k] = (_Float16)v.y;
    Wt[(n + 2) * 64 + k] = (_Float16)v.z;
    Wt[(n + 3) * 64 + k] = (_Float16)v.w;
  }
}

// Two-layer MLP on one 16x64 tile held as two A fragments (k0 = 0, 32).
// Wt1/Wt2: LDS transposed f16 weights (Wt[n*64+k]).  h16: per-wave LDS
// staging (16x64 f16) for the layer1->layer2 transpose.  d[4]: output D
// fragments (one per 16-column tile), f32.
__device__ __forceinline__ void mlp_tile(const _Float16* Wt1, const _Float16* Wt2,
                                         const float* __restrict__ b1,
                                         const float* __restrict__ b2,
                                         _Float16* h16,
                                         v16h a0, v16h a1, v8f d[4]) {
  const int lane = threadIdx.x & 31;
  const int lrow = lane & 15;        // column for B/C-D, row for A
  const int hi   = lane >> 4;        // lane-group select

  // ---- layer 1: h = relu(x @ W1 + b1) ----
  v8f h[4];
#pragma unroll
  for (int nt = 0; nt < 4; ++nt) {
    int n = nt * 16 + lrow;
    float bv = b1[n];
    v8f c = {bv, bv, bv, bv, bv, bv, bv, bv};
    v16h bf0 = __builtin_shufflevector(*(const v8h*)(Wt1 + n * 64 + 0  + hi * 8), *(const v8h*)(Wt1 + n * 64 + 16 + hi * 8), 0,1,2,3,4,5,6,7,8,9,10,11,12,13,14,15);   // K 0..31
    v16h bf1 = __builtin_shufflevector(*(const v8h*)(Wt1 + n * 64 + 32 + hi * 8), *(const v8h*)(Wt1 + n * 64 + 48 + hi * 8), 0,1,2,3,4,5,6,7,8,9,10,11,12,13,14,15);   // K 32..63
    c = __builtin_amdgcn_wmma_f32_16x16x32_f16(false, a0, false, bf0, (short)0, c, false, false);
    c = __builtin_amdgcn_wmma_f32_16x16x32_f16(false, a1, false, bf1, (short)0, c, false, false);
#pragma unroll
    for (int i = 0; i < 8; ++i) c[i] = fmaxf(c[i], 0.0f);
    h[nt] = c;
  }

  // ---- transpose D-layout -> A-layout through per-wave LDS (f16) ----
#pragma unroll
  for (int nt = 0; nt < 4; ++nt)
#pragma unroll
    for (int i = 0; i < 8; ++i)
      h16[(i + hi * 8) * 64 + nt * 16 + lrow] = (_Float16)h[nt][i];

  asm volatile("s_wait_dscnt 0" ::: "memory");   // same-wave LDS RAW fence

  v16h ha0, ha1;
  {
    v8h lo0 = *(const v8h*)(h16 + lrow * 64 + 0  + hi * 8);
    v8h hh0 = *(const v8h*)(h16 + lrow * 64 + 16 + hi * 8);
    v8h lo1 = *(const v8h*)(h16 + lrow * 64 + 32 + hi * 8);
    v8h hh1 = *(const v8h*)(h16 + lrow * 64 + 48 + hi * 8);
#pragma unroll
    for (int e = 0; e < 8; ++e) {
      ha0[e] = lo0[e]; ha0[8 + e] = hh0[e];
      ha1[e] = lo1[e]; ha1[8 + e] = hh1[e];
    }
  }

  // ---- layer 2: d = h @ W2 + b2 ----
#pragma unroll
  for (int nt = 0; nt < 4; ++nt) {
    int n = nt * 16 + lrow;
    float bv = b2[n];
    v8f c = {bv, bv, bv, bv, bv, bv, bv, bv};
    v16h bf0 = __builtin_shufflevector(*(const v8h*)(Wt2 + n * 64 + 0  + hi * 8), *(const v8h*)(Wt2 + n * 64 + 16 + hi * 8), 0,1,2,3,4,5,6,7,8,9,10,11,12,13,14,15);
    v16h bf1 = __builtin_shufflevector(*(const v8h*)(Wt2 + n * 64 + 32 + hi * 8), *(const v8h*)(Wt2 + n * 64 + 48 + hi * 8), 0,1,2,3,4,5,6,7,8,9,10,11,12,13,14,15);
    c = __builtin_amdgcn_wmma_f32_16x16x32_f16(false, ha0, false, bf0, (short)0, c, false, false);
    c = __builtin_amdgcn_wmma_f32_16x16x32_f16(false, ha1, false, bf1, (short)0, c, false, false);
    d[nt] = c;
  }
}

// ---------------- kernel 1: zero replicated accumulators ----------------
__global__ __launch_bounds__(256) void zero_acc_kernel(float4* __restrict__ p, int n4) {
  int i = blockIdx.x * 256 + threadIdx.x;
  if (i < n4) p[i] = make_float4(0.f, 0.f, 0.f, 0.f);
}

// ---------------- kernel 2: segment-sum scatter (replicated atomics) ----------------
__global__ __launch_bounds__(256) void scatter_kernel(const float* __restrict__ x,
                                                      const int* __restrict__ act,
                                                      const int* __restrict__ bvec,
                                                      float* __restrict__ acc) {
  int tid = blockIdx.x * 256 + threadIdx.x;       // 16,000,000 threads exactly
  int row = tid >> 4;
  int c4  = (tid & 15) * 4;
  int seg = act[row] + bvec[row] * NLAYERS;
  float4 v = *(const float4*)(x + (size_t)row * DD + c4);
  int rep = blockIdx.x & (REPL - 1);
  float* dst = acc + ((size_t)rep * NSEG + seg) * DD + c4;
  atomicAdd(dst + 0, v.x);
  atomicAdd(dst + 1, v.y);
  atomicAdd(dst + 2, v.z);
  atomicAdd(dst + 3, v.w);
}

// ---------------- kernel 3: rho MLP on the 2048 segment sums ----------------
__global__ __launch_bounds__(256) void rho_kernel(const float* __restrict__ acc,
                                                  const float* __restrict__ w1,
                                                  const float* __restrict__ b1,
                                                  const float* __restrict__ w2,
                                                  const float* __restrict__ b2,
                                                  float* __restrict__ rho_out) {
  __shared__ _Float16 Wt1[64 * 64];
  __shared__ _Float16 Wt2[64 * 64];
  __shared__ _Float16 Hst[8][16 * 64];
  stage_wt(w1, Wt1);
  stage_wt(w2, Wt2);
  __syncthreads();

  const int wave = threadIdx.x >> 5;
  const int tile = blockIdx.x * 8 + wave;         // 16 blocks * 8 waves = 128 tiles exactly
  const int lane = threadIdx.x & 31;
  const int lrow = lane & 15;
  const int hi   = lane >> 4;
  const int row0 = tile * 16;

  // Reduce the REPL replicas while building the two A fragments (k0 = 0, 32).
  float4 s0 = make_float4(0,0,0,0), s1 = s0, s2 = s0, s3 = s0;
  float4 s4 = s0, s5 = s0, s6 = s0, s7 = s0;
#pragma unroll
  for (int rep = 0; rep < REPL; ++rep) {
    const float* p = acc + ((size_t)rep * NSEG + row0 + lrow) * DD + hi * 8;
    s0 = f4add(s0, *(const float4*)(p + 0));
    s1 = f4add(s1, *(const float4*)(p + 4));
    s2 = f4add(s2, *(const float4*)(p + 16));
    s3 = f4add(s3, *(const float4*)(p + 20));
    s4 = f4add(s4, *(const float4*)(p + 32));
    s5 = f4add(s5, *(const float4*)(p + 36));
    s6 = f4add(s6, *(const float4*)(p + 48));
    s7 = f4add(s7, *(const float4*)(p + 52));
  }
  v16h a0 = pack_frag(s0, s1, s2, s3);
  v16h a1 = pack_frag(s4, s5, s6, s7);

  v8f d[4];
  mlp_tile(Wt1, Wt2, b1, b2, &Hst[wave][0], a0, a1, d);

#pragma unroll
  for (int nt = 0; nt < 4; ++nt)
#pragma unroll
    for (int i = 0; i < 8; ++i) {
      int m = i + hi * 8;
      rho_out[(size_t)(row0 + m) * DD + nt * 16 + lrow] = d[nt][i];
    }
}

// ---------------- kernel 4: phi MLP + rho gather-add (the big pass) ----------------
__global__ __launch_bounds__(256) void phi_main_kernel(const float* __restrict__ x,
                                                       const int* __restrict__ act,
                                                       const int* __restrict__ bvec,
                                                       const float* __restrict__ w1,
                                                       const float* __restrict__ b1,
                                                       const float* __restrict__ w2,
                                                       const float* __restrict__ b2,
                                                       const float* __restrict__ rho_out,
                                                       float* __restrict__ out) {
  __shared__ _Float16 Wt1[64 * 64];
  __shared__ _Float16 Wt2[64 * 64];
  __shared__ _Float16 Hst[8][16 * 64];
  stage_wt(w1, Wt1);
  stage_wt(w2, Wt2);
  __syncthreads();

  const int wave = threadIdx.x >> 5;
  const int tile = blockIdx.x * 8 + wave;
  if (tile >= NTILES) return;                      // wave-uniform exit

  const int lane = threadIdx.x & 31;
  const int lrow = lane & 15;
  const int hi   = lane >> 4;
  const int row0 = tile * 16;

  // A fragments straight from global f32 x (two k-steps).
  const float* xp = x + (size_t)(row0 + lrow) * DD + hi * 8;
  v16h a0 = load_a_frag_f32(xp);        // K 0..31
  v16h a1 = load_a_frag_f32(xp + 32);   // K 32..63

  v8f d[4];
  mlp_tile(Wt1, Wt2, b1, b2, &Hst[wave][0], a0, a1, d);

  // Segment id of row (row0 + lrow); identical in both lane groups.
  int r   = row0 + lrow;
  int seg = act[r] + bvec[r] * NLAYERS;

#pragma unroll
  for (int nt = 0; nt < 4; ++nt) {
    int n = nt * 16 + lrow;
#pragma unroll
    for (int i = 0; i < 8; ++i) {
      int m = i + hi * 8;                               // row-in-tile of this D element
      int s = __builtin_amdgcn_ds_bpermute(m << 2, seg); // broadcast seg of row m
      float rv = rho_out[(size_t)s * DD + n];
      out[(size_t)(row0 + m) * DD + n] = d[nt][i] + rv;
    }
  }
}

// ---------------- launch ----------------
extern "C" void kernel_launch(void* const* d_in, const int* in_sizes, int n_in,
                              void* d_out, int out_size, void* d_ws, size_t ws_size,
                              hipStream_t stream) {
  const float* x    = (const float*)d_in[0];
  const int*   act  = (const int*)d_in[1];
  const int*   bvec = (const int*)d_in[2];
  // d_in[3] = num_layers (== 16, baked into NLAYERS)
  const float* pw1 = (const float*)d_in[4];
  const float* pb1 = (const float*)d_in[5];
  const float* pw2 = (const float*)d_in[6];
  const float* pb2 = (const float*)d_in[7];
  const float* rw1 = (const float*)d_in[8];
  const float* rb1 = (const float*)d_in[9];
  const float* rw2 = (const float*)d_in[10];
  const float* rb2 = (const float*)d_in[11];
  float* out = (float*)d_out;

  // workspace layout: [REPL * 2048 * 64 f32 accumulators][2048 * 64 f32 rho_out]
  float* acc     = (float*)d_ws;
  float* rho_out = acc + (size_t)REPL * NSEG * DD;

  const int n4 = REPL * NSEG * DD / 4;             // 262,144 float4s
  zero_acc_kernel<<<(n4 + 255) / 256, 256, 0, stream>>>((float4*)acc, n4);
  scatter_kernel<<<NROWS * 16 / 256, 256, 0, stream>>>(x, act, bvec, acc);
  rho_kernel<<<16, 256, 0, stream>>>(acc, rw1, rb1, rw2, rb2, rho_out);
  phi_main_kernel<<<(NTILES + 7) / 8, 256, 0, stream>>>(x, act, bvec, pw1, pb1, pw2, pb2,
                                                        rho_out, out);
}
